// Attention_13769665151235
// MI455X (gfx1250) — matrix operand order repaired and checked by reading, not compiled, not run on hardware
//
#include <hip/hip_runtime.h>

// ---------------------------------------------------------------------------
// Types for CDNA5 WMMA (wave32): v_wmma_f32_16x16x32_bf16
// ---------------------------------------------------------------------------
typedef __bf16 bf16_t;
typedef __attribute__((ext_vector_type(16))) __bf16 v16bf;
typedef __attribute__((ext_vector_type(8)))  float  v8f;

union FragU {                 // 16 bf16 = 8 dwords = one WMMA A/B operand
    unsigned int u[8];
    v16bf v;
};
union PairU {                 // 2 bf16 <-> 1 dword
    unsigned int u;
    bf16_t h[2];
};

#define WMMA_BF16(a, b, c) \
    __builtin_amdgcn_wmma_f32_16x16x32_bf16(false, (a), false, (b), (short)0, (c), false, false)

// ---------------------------------------------------------------------------
// CDNA5 async global->LDS staging (GLOBAL_LOAD_ASYNC_TO_LDS_B128, ASYNCcnt).
// Builtin signature (from compiler diagnostic): takes int4-vector pointers,
// global (AS1) source and LDS (AS3) destination, plus imm offset + cpol.
// Guarded: falls back to VGPR round-trip copy if the builtin is absent.
// ---------------------------------------------------------------------------
#if defined(__has_builtin)
#  if __has_builtin(__builtin_amdgcn_global_load_async_to_lds_b128)
#    define HAVE_ASYNC_COPY 1
#  endif
#endif
#ifndef HAVE_ASYNC_COPY
#  define HAVE_ASYNC_COPY 0
#endif

typedef int v4i_vs __attribute__((vector_size(16)));   // int4, matches builtin param

__device__ __forceinline__ void copy_b128_to_lds(const void* g, void* l)
{
#if HAVE_ASYNC_COPY
    __builtin_amdgcn_global_load_async_to_lds_b128(
        (__attribute__((address_space(1))) v4i_vs*)g,
        (__attribute__((address_space(3))) v4i_vs*)l,
        0, 0);
#else
    *(uint4*)l = *(const uint4*)g;
#endif
}

__device__ __forceinline__ void async_join()
{
#if HAVE_ASYNC_COPY
    asm volatile("s_wait_asynccnt 0x0" ::: "memory");
#endif
}

// ---------------------------------------------------------------------------
// Tiled GEMM with bias:  Out[M,N] = A[M,K] * W[K,N] + bias[N]
//   A: f32 or bf16 (template), W/bias: f32, Out: bf16 or f32 (template)
//   Block: 256 thr = 8 waves (2x4), tile 128x128, BK=32.
//   Each wave: 4x2 grid of 16x16 WMMA tiles => 8 v_wmma per K-step.
// ---------------------------------------------------------------------------
template <bool A_F32, bool OUT_F32>
__global__ __launch_bounds__(256)
void gemm_bias_wmma(const void* __restrict__ Aptr,
                    const float* __restrict__ W,
                    const float* __restrict__ bias,
                    void* __restrict__ Out,
                    int M, int N, int K)
{
    constexpr int BM = 128, BN = 128, BK = 32;
    constexpr int LDA = BK + 8;   // bf16 elems, padded
    constexpr int LDB = BK + 8;   // B stored transposed [n][k]
    __shared__ bf16_t sA[BM * LDA];
    __shared__ bf16_t sB[BN * LDB];

    const int tid   = threadIdx.x;
    const int lane  = tid & 31;
    const int wave  = tid >> 5;
    const int waveM = wave & 1;   // 0..1 -> 64 rows each
    const int waveN = wave >> 1;  // 0..3 -> 32 cols each
    const int half  = lane >> 4;
    const int l16   = lane & 15;
    const int blockM = blockIdx.x * BM;
    const int blockN = blockIdx.y * BN;

    v8f acc[4][2];
    v8f zero = {};
#pragma unroll
    for (int mt = 0; mt < 4; ++mt)
#pragma unroll
        for (int nt = 0; nt < 2; ++nt) acc[mt][nt] = zero;

    for (int k0 = 0; k0 < K; k0 += BK) {
        // ---- stage A tile -> LDS [m][k] (bf16) ----
        if (A_F32) {
            const float* A = (const float*)Aptr;
#pragma unroll
            for (int i = 0; i < 4; ++i) {
                int idx = tid + i * 256;          // 1024 float4 slots
                int row = idx >> 3;
                int col = (idx & 7) * 4;
                const float4 f = *(const float4*)(A + (size_t)(blockM + row) * K + k0 + col);
                bf16_t t[4] = { (bf16_t)f.x, (bf16_t)f.y, (bf16_t)f.z, (bf16_t)f.w };
                *(uint2*)&sA[row * LDA + col] = *(const uint2*)t;
            }
        } else {
            const bf16_t* A = (const bf16_t*)Aptr;
#pragma unroll
            for (int i = 0; i < 2; ++i) {
                int idx = tid + i * 256;          // 512 slots of 8 bf16
                int row = idx >> 2;
                int col = (idx & 3) * 8;
                copy_b128_to_lds(A + (size_t)(blockM + row) * K + k0 + col,
                                 &sA[row * LDA + col]);
            }
        }
        // ---- stage W tile -> LDS transposed [n][k] (bf16) ----
#pragma unroll
        for (int i = 0; i < 8; ++i) {
            int idx = tid + i * 256;              // 2048 k-pairs
            int kp  = idx >> 7;                   // 0..15
            int n   = idx & 127;
            int gk  = k0 + kp * 2;
            PairU p;
            p.h[0] = (bf16_t)W[(size_t)gk * N + blockN + n];
            p.h[1] = (bf16_t)W[(size_t)(gk + 1) * N + blockN + n];
            *(unsigned int*)&sB[n * LDB + kp * 2] = p.u;
        }
        // ---- prefetch next K-step tiles into cache hierarchy ----
        if (k0 + BK < K) {
            if (A_F32)
                __builtin_prefetch((const float*)Aptr +
                                   (size_t)(blockM + (tid >> 1)) * K + (k0 + BK), 0, 1);
            __builtin_prefetch(W + (size_t)(k0 + BK + (tid >> 7)) * N + blockN + (tid & 127), 0, 1);
        }
        if (!A_F32) async_join();
        __syncthreads();

        // ---- fragments + WMMA ----
        FragU afrag[4], bfrag[2];
#pragma unroll
        for (int mt = 0; mt < 4; ++mt) {
            const bf16_t* base = &sA[(waveM * 64 + mt * 16 + l16) * LDA];
#pragma unroll
            for (int v = 0; v < 8; ++v) {
                int kk = 2 * v + 16 * half;
                afrag[mt].u[v] = *(const unsigned int*)(base + kk);
            }
        }
#pragma unroll
        for (int nt = 0; nt < 2; ++nt) {
            const bf16_t* base = &sB[(waveN * 32 + nt * 16 + l16) * LDB + 16 * half];
#pragma unroll
            for (int v = 0; v < 8; ++v)
                bfrag[nt].u[v] = *(const unsigned int*)(base + 2 * v);
        }
#pragma unroll
        for (int mt = 0; mt < 4; ++mt)
#pragma unroll
            for (int nt = 0; nt < 2; ++nt)
                acc[mt][nt] = WMMA_BF16(afrag[mt].v, bfrag[nt].v, acc[mt][nt]);
        __syncthreads();
    }

    // ---- epilogue: bias + store (C layout: lane=N, half*8+v = M) ----
#pragma unroll
    for (int mt = 0; mt < 4; ++mt) {
#pragma unroll
        for (int nt = 0; nt < 2; ++nt) {
            int col  = blockN + waveN * 32 + nt * 16 + l16;
            float bv = bias[col];
#pragma unroll
            for (int v = 0; v < 8; ++v) {
                int row = blockM + waveM * 64 + mt * 16 + 8 * half + v;
                float val = acc[mt][nt][v] + bv;
                if (OUT_F32)
                    ((float*)Out)[(size_t)row * N + col] = val;
                else
                    ((bf16_t*)Out)[(size_t)row * N + col] = (bf16_t)val;
            }
        }
    }
}

// ---------------------------------------------------------------------------
// Flash attention (no causal mask, per reference), bf16 WMMA, f32 softmax.
//   qkv: [B*S, 3*NX] bf16 ; out: [B*S, NX] bf16 (heads merged)
//   Block = 128 thr = 4 waves; one block per (b, h, 64-row query tile).
//   Each wave owns 16 query rows; loops over 16 key blocks of 64.
// ---------------------------------------------------------------------------
__global__ __launch_bounds__(128)
void flash_attn_wmma(const bf16_t* __restrict__ qkv,
                     bf16_t* __restrict__ outbuf)
{
    constexpr int S = 1024, NX = 1024, HD = 64, QB = 64, KB = 64, LD = 72;
    __shared__ bf16_t sQ[QB * LD];        // [q][d]
    __shared__ bf16_t sK[KB * LD];        // [key][d]   (== B-frag layout for QK^T)
    __shared__ bf16_t sV[HD * LD];        // [d][key]   (== B-frag layout for P*V)
    __shared__ bf16_t sP[4][16 * LD];     // per-wave P staging, [q][key]

    const int tid  = threadIdx.x;
    const int lane = tid & 31;
    const int wave = tid >> 5;
    const int half = lane >> 4;
    const int l16  = lane & 15;

    const int qb = blockIdx.x & 15;
    const int h  = (blockIdx.x >> 4) & 15;
    const int b  = blockIdx.x >> 8;

    const size_t rs = 3 * NX;             // qkv row stride (elems)
    const bf16_t* Qg = qkv + (size_t)(b * S) * rs + (size_t)h * HD;
    const bf16_t* Kg = Qg + NX;
    const bf16_t* Vg = Qg + 2 * NX;

    // ---- load Q tile once (async copy to LDS) ----
#pragma unroll
    for (int i = 0; i < 4; ++i) {
        int idx = tid + i * 128;          // 512 uint4 slots
        int row = idx >> 3, c = (idx & 7) * 8;
        copy_b128_to_lds(Qg + (size_t)(qb * QB + row) * rs + c, &sQ[row * LD + c]);
    }
    async_join();
    __syncthreads();

    // preload this wave's Q A-fragments (two K-chunks of 32)
    FragU qf[2];
#pragma unroll
    for (int kc = 0; kc < 2; ++kc) {
        const bf16_t* base = &sQ[(wave * 16 + l16) * LD + kc * 32];
#pragma unroll
        for (int v = 0; v < 8; ++v) {
            int kk = 2 * v + 16 * half;
            qf[kc].u[v] = *(const unsigned int*)(base + kk);
        }
    }

    v8f o[4];
    v8f zero = {};
#pragma unroll
    for (int dt = 0; dt < 4; ++dt) o[dt] = zero;
    float m_i[8], l_i[8];
#pragma unroll
    for (int v = 0; v < 8; ++v) { m_i[v] = -1e30f; l_i[v] = 0.0f; }

    for (int kb = 0; kb < S / KB; ++kb) {
        __syncthreads();                  // protect sK/sV reuse
        // ---- K tile [key][d] via async copy ----
#pragma unroll
        for (int i = 0; i < 4; ++i) {
            int idx = tid + i * 128;
            int row = idx >> 3, c = (idx & 7) * 8;
            copy_b128_to_lds(Kg + (size_t)(kb * KB + row) * rs + c, &sK[row * LD + c]);
        }
        // ---- V tile transposed -> [d][key] (manual: needs transpose) ----
#pragma unroll
        for (int i = 0; i < 16; ++i) {
            int idx = tid + i * 128;      // 2048 dwords: key=idx>>5, d=(idx&31)*2
            int key = idx >> 5, du = (idx & 31) * 2;
            PairU p;
            p.u = *(const unsigned int*)(Vg + (size_t)(kb * KB + key) * rs + du);
            sV[du * LD + key]       = p.h[0];
            sV[(du + 1) * LD + key] = p.h[1];
        }
        async_join();
        __syncthreads();

        // ---- S = Q K^T : 4 accumulators of 16x16 across 64 keys ----
        v8f s[4];
#pragma unroll
        for (int nt = 0; nt < 4; ++nt) s[nt] = zero;
#pragma unroll
        for (int kc = 0; kc < 2; ++kc) {
#pragma unroll
            for (int nt = 0; nt < 4; ++nt) {
                FragU kf;
                const bf16_t* base = &sK[(nt * 16 + l16) * LD + kc * 32 + 16 * half];
#pragma unroll
                for (int v = 0; v < 8; ++v)
                    kf.u[v] = *(const unsigned int*)(base + 2 * v);
                s[nt] = WMMA_BF16(qf[kc].v, kf.v, s[nt]);
            }
        }

        // ---- online softmax (scale 1/sqrt(64) = 0.125) ----
#pragma unroll
        for (int nt = 0; nt < 4; ++nt)
#pragma unroll
            for (int v = 0; v < 8; ++v) s[nt][v] *= 0.125f;

        float alpha[8];
#pragma unroll
        for (int v = 0; v < 8; ++v) {
            float mx = fmaxf(fmaxf(s[0][v], s[1][v]), fmaxf(s[2][v], s[3][v]));
#pragma unroll
            for (int off = 8; off > 0; off >>= 1)
                mx = fmaxf(mx, __shfl_xor(mx, off, 16));   // reduce across N-lanes of this row
            float mn = fmaxf(m_i[v], mx);
            alpha[v] = __expf(m_i[v] - mn);
            float rsum = 0.0f;
#pragma unroll
            for (int nt = 0; nt < 4; ++nt) {
                float p = __expf(s[nt][v] - mn);
                s[nt][v] = p;
                rsum += p;
            }
#pragma unroll
            for (int off = 8; off > 0; off >>= 1)
                rsum += __shfl_xor(rsum, off, 16);
            l_i[v] = l_i[v] * alpha[v] + rsum;
            m_i[v] = mn;
        }
#pragma unroll
        for (int dt = 0; dt < 4; ++dt)
#pragma unroll
            for (int v = 0; v < 8; ++v) o[dt][v] *= alpha[v];

        // ---- stage P (bf16) per-wave, C-layout -> [q][key] in LDS ----
        bf16_t* pbuf = &sP[wave][0];
#pragma unroll
        for (int nt = 0; nt < 4; ++nt)
#pragma unroll
            for (int v = 0; v < 8; ++v)
                pbuf[(8 * half + v) * LD + nt * 16 + l16] = (bf16_t)s[nt][v];

        // ---- O += P * V ----
#pragma unroll
        for (int kc = 0; kc < 2; ++kc) {
            FragU pf;
            const bf16_t* base = &pbuf[l16 * LD + kc * 32];
#pragma unroll
            for (int v = 0; v < 8; ++v) {
                int kk = 2 * v + 16 * half;
                pf.u[v] = *(const unsigned int*)(base + kk);
            }
#pragma unroll
            for (int dt = 0; dt < 4; ++dt) {
                FragU vf;
                const bf16_t* vb = &sV[(dt * 16 + l16) * LD + kc * 32 + 16 * half];
#pragma unroll
                for (int v = 0; v < 8; ++v)
                    vf.u[v] = *(const unsigned int*)(vb + 2 * v);
                o[dt] = WMMA_BF16(pf.v, vf.v, o[dt]);
            }
        }
    }

    // ---- normalize + store merged-heads [B,S,NX] bf16 ----
    bf16_t* Og = outbuf + (size_t)(b * S + qb * QB) * NX + (size_t)h * HD;
#pragma unroll
    for (int dt = 0; dt < 4; ++dt) {
        int d = dt * 16 + l16;
#pragma unroll
        for (int v = 0; v < 8; ++v) {
            int row = wave * 16 + 8 * half + v;
            Og[(size_t)row * NX + d] = (bf16_t)(o[dt][v] * __frcp_rn(l_i[v]));
        }
    }
}

// ---------------------------------------------------------------------------
// Launcher
// ---------------------------------------------------------------------------
extern "C" void kernel_launch(void* const* d_in, const int* in_sizes, int n_in,
                              void* d_out, int out_size, void* d_ws, size_t ws_size,
                              hipStream_t stream)
{
    (void)in_sizes; (void)n_in; (void)out_size; (void)ws_size;
    constexpr int B = 4, S = 1024, NX = 1024;
    constexpr int M = B * S;            // 4096

    const float* hidden   = (const float*)d_in[0];
    const float* c_attn_w = (const float*)d_in[1];   // [NX, 3NX]
    const float* c_attn_b = (const float*)d_in[2];
    const float* c_proj_w = (const float*)d_in[3];   // [NX, NX]
    const float* c_proj_b = (const float*)d_in[4];
    float* out = (float*)d_out;

    char* ws = (char*)d_ws;
    bf16_t* qkv_bf16  = (bf16_t*)ws;                                   // 25.2 MB
    bf16_t* attn_bf16 = (bf16_t*)(ws + (size_t)M * 3 * NX * sizeof(bf16_t)); // 8.4 MB

    // 1) QKV projection: f32 in -> bf16 qkv
    {
        dim3 grid(M / 128, (3 * NX) / 128);
        gemm_bias_wmma<true, false><<<grid, 256, 0, stream>>>(
            hidden, c_attn_w, c_attn_b, qkv_bf16, M, 3 * NX, NX);
    }
    // 2) Flash attention over 16 heads, merged-head bf16 output
    {
        dim3 grid(B * 16 * (S / 64));   // b*256 + h*16 + qtile
        flash_attn_wmma<<<grid, 128, 0, stream>>>(qkv_bf16, attn_bf16);
    }
    // 3) Output projection: bf16 in -> f32 out
    {
        dim3 grid(M / 128, NX / 128);
        gemm_bias_wmma<false, true><<<grid, 256, 0, stream>>>(
            attn_bf16, c_proj_w, c_proj_b, out, M, NX, NX);
    }
}
